// Temporal_Learning_12575664242978
// MI455X (gfx1250) — matrix operand order repaired and checked by reading, not compiled, not run on hardware
//
#include <hip/hip_runtime.h>
#include <hip/hip_bf16.h>

// ---------------------------------------------------------------------------
// 2-layer LSTM (B=128, T=1024, H=512, IN=1) + linear head, persistent kernel.
// 32 workgroups x 256 threads (8 wave32). Each WG owns a 16-hidden-unit slice:
//   - its 3 weight slices (64x512 f16 each) live in LDS for the whole run
//   - c1/c2 cell state lives in VGPRs (v8f per wave, WMMA C/D layout)
//   - h1/h2 are double-buffered f16 (128x512) in global (L2-resident)
// Grid-wide sync via monotonic atomic-counter barrier (2 per timestep).
// ---------------------------------------------------------------------------

typedef _Float16 v8h  __attribute__((ext_vector_type(8)));
typedef _Float16 v16h __attribute__((ext_vector_type(16)));
typedef float    v8f  __attribute__((ext_vector_type(8)));

#define B_     128
#define T_     1024
#define H_     512
#define NW_    32      // workgroups (one per WGP)
#define SLICE_ 16      // hidden units per WG
#define NTHR_  256     // 8 wave32

static __device__ __forceinline__ v8f vzero8() { v8f z = {}; return z; }

static __device__ __forceinline__ float sig_(float x) {
  return 1.0f / (1.0f + __expf(-x));
}
static __device__ __forceinline__ float tanh_(float x) {
  float e = __expf(-2.0f * x);
  return (1.0f - e) / (1.0f + e);
}

// 16-bit A fragment (16x32): lane l<16 = row l, K {kb..kb+7, kb+16..kb+23};
// lanes 16-31 same rows, kb shifted by 8 (handled by caller via kselA).
static __device__ __forceinline__ v16h load_a_frag(const _Float16* p) {
  v8h lo = *(const v8h*)(p);
  v8h hi = *(const v8h*)(p + 16);
  return __builtin_shufflevector(lo, hi, 0, 1, 2, 3, 4, 5, 6, 7,
                                         8, 9, 10, 11, 12, 13, 14, 15);
}

static __device__ __forceinline__ v8f wmma_(v16h a, v16h b, v8f c) {
  // D = A(16x32) * B(32x16) + C ; emits v_wmma_f32_16x16x32_f16
  return __builtin_amdgcn_wmma_f32_16x16x32_f16(
      false, a, false, b, (short)0, c, false, false);
}

// One (128 x 64)-output GEMM slice contribution, K = 512 (16 k-steps).
// hsrc: f16 h matrix (128 x 512) in global. sW: 64x512 f16 weight slice in LDS
// (row = gate*16 + n). B fragment: lane = column n, K {kb..kb+7, kb+16..kb+23}.
static __device__ __forceinline__ void gemm_acc(
    const _Float16* __restrict__ hsrc, const _Float16* sW,
    int rowA, int kselA, int kselB, int l15, v8f acc[4]) {
#pragma unroll 2
  for (int ks = 0; ks < 16; ++ks) {
    const int k0 = ks * 32;
    v16h a = load_a_frag(hsrc + rowA * H_ + k0 + kselA);
#pragma unroll
    for (int g = 0; g < 4; ++g) {
      v16h b = load_a_frag(sW + (g * 16 + l15) * H_ + k0 + kselB);
      acc[g] = wmma_(a, b, acc[g]);
    }
  }
}

static __device__ __forceinline__ void grid_barrier(unsigned* ctr,
                                                    unsigned target) {
  __syncthreads();
  if (threadIdx.x == 0) {
    __threadfence();
    atomicAdd(ctr, 1u);
    while (__hip_atomic_load(ctr, __ATOMIC_ACQUIRE,
                             __HIP_MEMORY_SCOPE_AGENT) < target)
      __builtin_amdgcn_s_sleep(2);
  }
  __syncthreads();
  __threadfence();
}

// Init: out = blin (linear-head bias; partial dots atomically accumulate),
// zero both h double-buffers, zero the barrier counter. Re-run every launch.
__global__ void lstm_init_kernel(float* __restrict__ out,
                                 const float* __restrict__ blin,
                                 unsigned* __restrict__ ctr,
                                 unsigned* __restrict__ hwords) {
  const int i = blockIdx.x * blockDim.x + threadIdx.x;
  if (i < B_ * T_) out[i] = blin[0];
  // h region: 2 layers * 2 buffers * B*H halfs = 524288 B = 131072 words
  if (i < (2 * 2 * B_ * H_ * 2) / 4) hwords[i] = 0u;
  if (i == 0) *ctr = 0u;
}

__global__ void __launch_bounds__(NTHR_, 1)
lstm2_persistent(const float* __restrict__ x,
                 const float* __restrict__ Wih1,
                 const float* __restrict__ Whh1,
                 const float* __restrict__ bih1,
                 const float* __restrict__ bhh1,
                 const float* __restrict__ Wih2,
                 const float* __restrict__ Whh2,
                 const float* __restrict__ bih2,
                 const float* __restrict__ bhh2,
                 const float* __restrict__ Wlin,
                 _Float16* __restrict__ h1buf,   // 2 * B*H halfs
                 _Float16* __restrict__ h2buf,   // 2 * B*H halfs
                 unsigned* __restrict__ ctr,
                 float* __restrict__ out) {
  extern __shared__ _Float16 smem[];
  _Float16* sW1  = smem;               // Whh1 slice  (64 x 512)
  _Float16* sW2i = smem + 64 * H_;     // Wih2 slice
  _Float16* sW2h = smem + 2 * 64 * H_; // Whh2 slice

  const int wg   = blockIdx.x;   // 0..31 : hidden slice owner
  const int tid  = threadIdx.x;
  const int lane = tid & 31;
  const int wv   = tid >> 5;     // wave 0..7 -> batch row-tile
  const int l15  = lane & 15;
  const int hi   = lane >> 4;

  // ---- stage this WG's weight slices into LDS (f32 -> f16), once ----
  for (int idx = tid; idx < 64 * H_; idx += NTHR_) {
    const int q = idx >> 9, k = idx & (H_ - 1);
    const int grow = (q >> 4) * H_ + wg * SLICE_ + (q & 15); // gate*H + j
    sW1 [idx] = (_Float16)Whh1[grow * H_ + k];
    sW2i[idx] = (_Float16)Wih2[grow * H_ + k];
    sW2h[idx] = (_Float16)Whh2[grow * H_ + k];
  }
  __syncthreads();

  // ---- per-lane constants (hidden unit j owned by this lane column) ----
  const int j = wg * SLICE_ + l15;
  float b1v[4], b2v[4], w1x[4];
#pragma unroll
  for (int g = 0; g < 4; ++g) {
    b1v[g] = bih1[g * H_ + j] + bhh1[g * H_ + j];
    b2v[g] = bih2[g * H_ + j] + bhh2[g * H_ + j];
    w1x[g] = Wih1[g * H_ + j];  // IN == 1
  }
  const float wlv = Wlin[j];

  const int rowA  = wv * 16 + l15;  // A-fragment row this lane loads
  const int kselA = hi * 8;         // A k sub-offset per lane half
  const int kselB = hi * 8;         // B k sub-offset per lane half
  const int bbase = wv * 16 + hi * 8;  // C/D: batch row = bbase + vgpr r

  v8f c1s = vzero8(), c2s = vzero8();  // persistent cell state in VGPRs
  int p = 0;
  unsigned bar = 0;

  for (int t = 0; t < T_; ++t) {
    const _Float16* h1r = h1buf + p * (B_ * H_);
    _Float16*       h1w = h1buf + (p ^ 1) * (B_ * H_);
    const _Float16* h2r = h2buf + p * (B_ * H_);
    _Float16*       h2w = h2buf + (p ^ 1) * (B_ * H_);

    float xv[8];
#pragma unroll
    for (int r = 0; r < 8; ++r) xv[r] = x[(bbase + r) * T_ + t];

    // ---------------- layer 1: gates = h1 @ Whh1^T + x*wih1 + b1 ----------
    v8f acc[4];
#pragma unroll
    for (int g = 0; g < 4; ++g) acc[g] = vzero8();
    gemm_acc(h1r, sW1, rowA, kselA, kselB, l15, acc);

#pragma unroll
    for (int r = 0; r < 8; ++r) {
      const float gi = sig_ (acc[0][r] + b1v[0] + w1x[0] * xv[r]);
      const float gf = sig_ (acc[1][r] + b1v[1] + w1x[1] * xv[r]);
      const float gg = tanh_(acc[2][r] + b1v[2] + w1x[2] * xv[r]);
      const float go = sig_ (acc[3][r] + b1v[3] + w1x[3] * xv[r]);
      const float c  = gf * c1s[r] + gi * gg;
      c1s[r] = c;
      h1w[(bbase + r) * H_ + j] = (_Float16)(go * tanh_(c));
    }
    bar += NW_;
    grid_barrier(ctr, bar);

    // ---------------- layer 2: gates = h1n @ Wih2^T + h2 @ Whh2^T + b2 ----
#pragma unroll
    for (int g = 0; g < 4; ++g) acc[g] = vzero8();
    gemm_acc(h1w, sW2i, rowA, kselA, kselB, l15, acc);
    gemm_acc(h2r, sW2h, rowA, kselA, kselB, l15, acc);

#pragma unroll
    for (int r = 0; r < 8; ++r) {
      const float gi = sig_ (acc[0][r] + b2v[0]);
      const float gf = sig_ (acc[1][r] + b2v[1]);
      const float gg = tanh_(acc[2][r] + b2v[2]);
      const float go = sig_ (acc[3][r] + b2v[3]);
      const float c  = gf * c2s[r] + gi * gg;
      c2s[r] = c;
      const float h = go * tanh_(c);
      h2w[(bbase + r) * H_ + j] = (_Float16)h;
      // linear head partial: reduce 16 hidden units of this slice per batch b
      float v = h * wlv;
      v += __shfl_xor(v, 1);
      v += __shfl_xor(v, 2);
      v += __shfl_xor(v, 4);
      v += __shfl_xor(v, 8);
      if (l15 == 0) atomicAdd(&out[(bbase + r) * T_ + t], v);
    }
    bar += NW_;
    grid_barrier(ctr, bar);
    p ^= 1;
  }
}

extern "C" void kernel_launch(void* const* d_in, const int* in_sizes, int n_in,
                              void* d_out, int out_size, void* d_ws,
                              size_t ws_size, hipStream_t stream) {
  (void)in_sizes; (void)n_in; (void)out_size; (void)ws_size;
  const float* x    = (const float*)d_in[0];
  const float* Wih1 = (const float*)d_in[1];
  const float* Whh1 = (const float*)d_in[2];
  const float* bih1 = (const float*)d_in[3];
  const float* bhh1 = (const float*)d_in[4];
  const float* Wih2 = (const float*)d_in[5];
  const float* Whh2 = (const float*)d_in[6];
  const float* bih2 = (const float*)d_in[7];
  const float* bhh2 = (const float*)d_in[8];
  const float* Wlin = (const float*)d_in[9];
  const float* blin = (const float*)d_in[10];
  float* out = (float*)d_out;

  unsigned char* ws = (unsigned char*)d_ws;
  unsigned*  ctr   = (unsigned*)ws;                                 // 256 B
  _Float16*  h1buf = (_Float16*)(ws + 256);                         // 256 KB
  _Float16*  h2buf = (_Float16*)(ws + 256 + 2 * B_ * H_ * sizeof(_Float16));

  const int initN = B_ * T_;  // also == h-region word count (131072)
  lstm_init_kernel<<<(initN + 255) / 256, 256, 0, stream>>>(
      out, blin, ctr, (unsigned*)(ws + 256));

  const size_t lds_bytes = 3u * 64u * H_ * sizeof(_Float16);  // 192 KB
  lstm2_persistent<<<NW_, NTHR_, lds_bytes, stream>>>(
      x, Wih1, Whh1, bih1, bhh1, Wih2, Whh2, bih2, bhh2, Wlin,
      h1buf, h2buf, ctr, out);
}
